// DeepMAR_ResNet50_42004780155503
// MI455X (gfx1250) — matrix operand order repaired and checked by reading, not compiled, not run on hardware
//
#include <hip/hip_runtime.h>

// ---------------------------------------------------------------------------
// Types & helpers
// ---------------------------------------------------------------------------
typedef __attribute__((ext_vector_type(16))) __bf16        v16bf;
typedef __attribute__((ext_vector_type(8)))  float         v8f;
typedef __attribute__((ext_vector_type(8)))  unsigned int  v8u;
typedef __attribute__((ext_vector_type(4)))  unsigned int  u32x4;
typedef __attribute__((ext_vector_type(8)))  int           i32x8;
typedef __attribute__((ext_vector_type(4)))  int           i32x4;

struct U32x8 { u32x4 lo; u32x4 hi; };  // 32 bytes, bit-castable to v16bf

#define DEV __device__ __forceinline__

DEV unsigned short f2bf(float f) {
  unsigned u = __builtin_bit_cast(unsigned, f);
  return (unsigned short)((u + 0x7FFFu + ((u >> 16) & 1u)) >> 16);  // RNE
}
DEV float bf2f(unsigned short h) {
  unsigned u = ((unsigned)h) << 16;
  return __builtin_bit_cast(float, u);
}

// ---------------------------------------------------------------------------
// Small prep kernels
// ---------------------------------------------------------------------------
__global__ void pack_input_kernel(const float* __restrict__ x,
                                  unsigned short* __restrict__ dst, int total) {
  int idx = blockIdx.x * 256 + threadIdx.x;
  if (idx >= total) return;
  int c = idx & 3;
  int p = idx >> 2;
  int w = p % 224; p /= 224;
  int h = p % 224;
  int n = p / 224;
  float v = (c < 3) ? x[(((long long)n * 3 + c) * 224 + h) * 224 + w] : 0.f;
  dst[idx] = f2bf(v);
}

// OIHW fp32 -> [OC][KH][KWp][ICp] bf16 (zero-padded phantom kw / ic)
__global__ void conv_w_prep_kernel(const float* __restrict__ src,
                                   unsigned short* __restrict__ dst,
                                   int srcIC, int KH, int KW, int KWp, int ICp,
                                   long long total) {
  long long idx = (long long)blockIdx.x * 256 + threadIdx.x;
  if (idx >= total) return;
  int ic = (int)(idx % ICp);
  long long r = idx / ICp;
  int kw = (int)(r % KWp); r /= KWp;
  int kh = (int)(r % KH);
  long long oc = r / KH;
  float v = 0.f;
  if (kw < KW && ic < srcIC)
    v = src[((oc * srcIC + ic) * KH + kh) * KW + kw];
  dst[idx] = f2bf(v);
}

__global__ void bn_prep_kernel(const float* g, const float* b, const float* m,
                               const float* v, float* sc, float* bi, int n) {
  int i = blockIdx.x * 256 + threadIdx.x;
  if (i >= n) return;
  float s = g[i] * rsqrtf(v[i] + 1e-5f);
  sc[i] = s;
  bi[i] = b[i] - m[i] * s;
}

// [A][K][N] fp32 -> [A][N][K] bf16 (GEMM B operand, row = N, contiguous K)
__global__ void gemm_w_prep_kernel(const float* __restrict__ src,
                                   unsigned short* __restrict__ dst,
                                   int K, int N, long long total) {
  long long idx = (long long)blockIdx.x * 256 + threadIdx.x;
  if (idx >= total) return;
  int k = (int)(idx % K);
  long long r = idx / K;
  int n = (int)(r % N);
  long long a = r / N;
  dst[idx] = f2bf(src[(a * K + k) * N + n]);
}

__global__ void cvt_bf16_kernel(const float* __restrict__ s,
                                unsigned short* __restrict__ d, long long n) {
  long long i = (long long)blockIdx.x * 256 + threadIdx.x;
  if (i < n) d[i] = f2bf(s[i]);
}

// ---------------------------------------------------------------------------
// Implicit-GEMM convolution with WMMA + double-buffered TDM weight staging.
// Activations NHWC bf16 (C = power of two). Weights [OC][KH*KWp*ICp] bf16.
// Block: 256 threads (8 waves). Tile: M=128 (output pixels) x N=64 (OC).
// Wave (4x2 grid) owns 32x32 via 2x2 v_wmma_f32_16x16x32_bf16 fragments.
//
// GENERIC=false (all bottleneck convs): IC % 32 == 0, so each 32-wide K-step
// stays inside one (kh,kw) filter cell -> cell state tracked as wave-uniform
// scalars, one bounds check + two b128 loads per A fragment.
// GENERIC=true (stem, IC=4, KWp=8): per-element K decode, all shifts/masks.
// RES / RELU: compile-time epilogue fusion (residual add / ReLU).
// ---------------------------------------------------------------------------
struct ConvP {
  const unsigned short* inp;
  const unsigned short* wgt;
  const float* scale;
  const float* bias;
  const unsigned short* res;   // residual (bf16 NHWC, OC); used iff RES
  unsigned short* outp;
  int IH, IW, icShift, OH, OW, OC, KWp, kwShift, Ktot, stride, pad;
};

template <bool GENERIC, bool RES, bool RELU>
__global__ __launch_bounds__(256) void conv_igemm_wmma(ConvP a) {
  __shared__ alignas(32) unsigned short ldsB[2][64 * 32];  // 2 x 4 KB

  const int tid   = threadIdx.x;
  const int wave  = tid >> 5;
  const int lane  = tid & 31;
  const int l16   = lane & 15;
  const int khalf = lane >> 4;
  const int waveM = wave >> 1;   // 0..3
  const int waveN = wave & 1;    // 0..1
  const int nBlock = blockIdx.x * 64;
  const int mBlock = blockIdx.y * 128;
  const int M      = 32 * a.OH * a.OW;
  const int icMask = (1 << a.icShift) - 1;
  const int IC     = 1 << a.icShift;
  const int kwMask = a.KWp - 1;  // KWp is a power of two in the GENERIC path

  // Per-lane output-pixel coordinates for the two A subtiles (rows = l16).
  int bcoord[2], ohc[2], owc[2];
#pragma unroll
  for (int s = 0; s < 2; s++) {
    int m = mBlock + waveM * 32 + s * 16 + l16;
    if (m >= M) m = M - 1;  // clamped load; store is guarded separately
    int ohw = a.OH * a.OW;
    bcoord[s] = m / ohw;
    int r = m - bcoord[s] * ohw;
    ohc[s] = r / a.OW;
    owc[s] = r - ohc[s] * a.OW;
  }

  v8f acc[2][2] = {};
  const int nIter = a.Ktot >> 5;

  // Wave-uniform filter-cell cursor (fast path).
  int icb = 0, khc = 0, kwc = 0;

  // One K-step of the MAC pipeline: A gather + LDS B fragments + 4 WMMAs.
  auto do_step = [&](int it) {
    const int k0 = it << 5;
    const int buf = it & 1;

    // ---- A fragments ------------------------------------------------------
    v16bf afrag[2];
    if constexpr (!GENERIC) {
#pragma unroll
      for (int s = 0; s < 2; s++) {
        int ih = ohc[s] * a.stride - a.pad + khc;
        int iw = owc[s] * a.stride - a.pad + kwc;
        bool ok = (ih >= 0) & (ih < a.IH) & (iw >= 0) & (iw < a.IW);
        const unsigned short* p = a.inp +
            ((((long long)bcoord[s] * a.IH + ih) * a.IW + iw) << a.icShift) +
            icb + khalf * 8;
        U32x8 w;
        w.lo = u32x4{0u, 0u, 0u, 0u};
        w.hi = u32x4{0u, 0u, 0u, 0u};
        if (ok) {
          w.lo = *(const u32x4*)p;         // K = khalf*8 + 0..7
          w.hi = *(const u32x4*)(p + 16);  // K = 16 + khalf*8 + 0..7
          if (s == 0) __builtin_prefetch(p + IC, 0, 1);
        }
        afrag[s] = __builtin_bit_cast(v16bf, w);
      }
    } else {
#pragma unroll
      for (int s = 0; s < 2; s++) {
        v8u w = { 0, 0, 0, 0, 0, 0, 0, 0 };
#pragma unroll
        for (int i = 0; i < 8; i++) {
          int kl = (i < 4) ? (khalf * 8 + 2 * i)
                           : (16 + khalf * 8 + 2 * (i - 4));
          int k = k0 + kl;
          int cell = k >> a.icShift;
          int ic = k & icMask;
          int kh = cell >> a.kwShift;    // KWp is a power of two
          int kw = cell & kwMask;
          int ih = ohc[s] * a.stride - a.pad + kh;
          int iw = owc[s] * a.stride - a.pad + kw;
          unsigned val = 0u;
          if (ih >= 0 && ih < a.IH && iw >= 0 && iw < a.IW)
            val = *(const unsigned*)(a.inp +
                  ((((long long)bcoord[s] * a.IH + ih) * a.IW + iw)
                   << a.icShift) + ic);
          w[i] = val;
        }
        afrag[s] = __builtin_bit_cast(v16bf, w);
      }
    }

    // ---- B fragments from LDS (K = khalf*8 + 0..7, 16 + khalf*8 + 0..7) ---
    v16bf bfrag[2];
#pragma unroll
    for (int t = 0; t < 2; t++) {
      int nl = waveN * 32 + t * 16 + l16;
      const unsigned short* bp = &ldsB[buf][nl * 32 + khalf * 8];
      U32x8 w;
      w.lo = *(const u32x4*)bp;
      w.hi = *(const u32x4*)(bp + 16);
      bfrag[t] = __builtin_bit_cast(v16bf, w);
    }

#pragma unroll
    for (int s = 0; s < 2; s++)
#pragma unroll
      for (int t = 0; t < 2; t++)
        acc[s][t] = __builtin_amdgcn_wmma_f32_16x16x32_bf16(
            false, afrag[s], false, bfrag[t], (short)0, acc[s][t], false, false);

    // advance wave-uniform filter-cell cursor
    if constexpr (!GENERIC) {
      icb += 32;
      if (icb >= IC) {
        icb = 0;
        kwc++;
        if (kwc >= a.KWp) { kwc = 0; khc++; }
      }
    }
  };

#if __has_builtin(__builtin_amdgcn_tensor_load_to_lds)
  // One wave drives the Tensor Data Mover; D# groups live in SGPRs.
  auto tdm_issue = [&](int k0, int buf) {
    unsigned long long ga =
        (unsigned long long)(const void*)a.wgt +
        (((unsigned long long)(unsigned)nBlock * (unsigned)a.Ktot +
          (unsigned)k0) << 1);
    unsigned ldsa = (unsigned)(unsigned long long)(const void*)ldsB[buf];
    // group0: count=1 | lds_addr | global_addr[56:0] | type=2
    u32x4 g0 = { 1u, ldsa, (unsigned)ga,
                 ((unsigned)((ga >> 32) & 0x01FFFFFFull)) | 0x80000000u };
    unsigned td0 = (unsigned)a.Ktot;  // contiguous K dimension
    unsigned td1 = (unsigned)a.OC;    // OC rows
    // group1: data_size=2B; tile = 32(K) x 64(rows); dim0 stride = Ktot
    i32x8 g1 = { (int)0x00010000u,
                 (int)((td0 & 0xFFFFu) << 16),
                 (int)((td0 >> 16) | ((td1 & 0xFFFFu) << 16)),
                 (int)((td1 >> 16) | (32u << 16)),
                 (int)64u,
                 (int)td0,
                 0, 0 };
    i32x4 z4 = { 0, 0, 0, 0 };
#if defined(__clang_major__) && __clang_major__ >= 23
    i32x8 z8 = { 0, 0, 0, 0, 0, 0, 0, 0 };
    __builtin_amdgcn_tensor_load_to_lds(g0, g1, z4, z4, z8, 0);
#else
    __builtin_amdgcn_tensor_load_to_lds(g0, g1, z4, z4, 0);
#endif
  };

  if (wave == 0) tdm_issue(0, 0);
  // Main loop: unconditionally issue iteration it+1's DMA, keep one in
  // flight (tensorcnt <= 1), compute iteration it. Last step peeled.
  for (int it = 0; it < nIter - 1; ++it) {
    if (wave == 0) {
      tdm_issue((it + 1) << 5, (it + 1) & 1);
      __builtin_amdgcn_s_wait_tensorcnt(1);
    }
    __syncthreads();
    do_step(it);
    __syncthreads();
  }
  if (wave == 0) __builtin_amdgcn_s_wait_tensorcnt(0);
  __syncthreads();
  do_step(nIter - 1);
#else
  for (int it = 0; it < nIter; ++it) {
    const int k0 = it << 5;
    const int buf = it & 1;
    for (int idx = tid; idx < 64 * 16; idx += 256) {
      int row = idx >> 4, kd = (idx & 15) * 2;
      ((unsigned*)ldsB[buf])[idx] = *(const unsigned*)(
          a.wgt + (long long)(nBlock + row) * a.Ktot + k0 + kd);
    }
    __syncthreads();
    do_step(it);
    __syncthreads();
  }
#endif

  // ---- epilogue: BN scale/bias (+residual) (+ReLU), store bf16 NHWC -------
  const bool fullM = (mBlock + 128 <= M);  // whole-block uniform fast path
#pragma unroll
  for (int s = 0; s < 2; s++)
#pragma unroll
    for (int t = 0; t < 2; t++) {
      int n = nBlock + waveN * 32 + t * 16 + l16;
      float sc = a.scale[n], bi = a.bias[n];
      int mBase = mBlock + waveM * 32 + s * 16 + khalf * 8;
      if (fullM) {
#pragma unroll
        for (int r = 0; r < 8; r++) {
          int m = mBase + r;
          float v = acc[s][t][r] * sc + bi;
          if constexpr (RES) v += bf2f(a.res[(long long)m * a.OC + n]);
          if constexpr (RELU) v = fmaxf(v, 0.f);
          a.outp[(long long)m * a.OC + n] = f2bf(v);
        }
      } else {
#pragma unroll
        for (int r = 0; r < 8; r++) {
          int m = mBase + r;
          if (m < M) {
            float v = acc[s][t][r] * sc + bi;
            if constexpr (RES) v += bf2f(a.res[(long long)m * a.OC + n]);
            if constexpr (RELU) v = fmaxf(v, 0.f);
            a.outp[(long long)m * a.OC + n] = f2bf(v);
          }
        }
      }
    }
}

// ---------------------------------------------------------------------------
// Batched GEMM, M fixed at 32: C[z] = act(A[z][32xK] * B[z][NxK]^T + bias)
// One block = 8 waves x 32 N columns = 256 N per block. grid=(ceil(N/256), Z)
// ACT: 0 = none, 1 = ReLU, 2 = leaky-ReLU(0.2) (compile-time, branchless).
// ---------------------------------------------------------------------------
template <int ACT>
__global__ __launch_bounds__(256) void gemm_bf16_m32(
    const unsigned short* __restrict__ Am, long long As,
    const unsigned short* __restrict__ Bm, long long Bs,
    const float* __restrict__ bias, long long biasS,
    float* __restrict__ Cm, long long Cs, int K, int N, int) {
  const int tid = threadIdx.x;
  const int wave = tid >> 5, lane = tid & 31, l16 = lane & 15, khalf = lane >> 4;
  const int z = blockIdx.y;
  const int nb = blockIdx.x * 256 + wave * 32;
  const unsigned short* A = Am + (long long)z * As;
  const unsigned short* B = Bm + (long long)z * Bs;

  v8f acc[2][2] = {};
  for (int k0 = 0; k0 < K; k0 += 32) {
    v16bf afrag[2], bfrag[2];
#pragma unroll
    for (int s = 0; s < 2; s++) {
      const unsigned short* p = A + (long long)(s * 16 + l16) * K + k0 + khalf * 8;
      U32x8 w;
      w.lo = *(const u32x4*)p;         // K = khalf*8 + 0..7
      w.hi = *(const u32x4*)(p + 16);  // K = 16 + khalf*8 + 0..7
      afrag[s] = __builtin_bit_cast(v16bf, w);
    }
#pragma unroll
    for (int t = 0; t < 2; t++) {
      int n = nb + t * 16 + l16;
      if (n > N - 1) n = N - 1;
      const unsigned short* p = B + (long long)n * K + k0 + khalf * 8;
      U32x8 w;
      w.lo = *(const u32x4*)p;
      w.hi = *(const u32x4*)(p + 16);
      bfrag[t] = __builtin_bit_cast(v16bf, w);
    }
#pragma unroll
    for (int s = 0; s < 2; s++)
#pragma unroll
      for (int t = 0; t < 2; t++)
        acc[s][t] = __builtin_amdgcn_wmma_f32_16x16x32_bf16(
            false, afrag[s], false, bfrag[t], (short)0, acc[s][t], false, false);
  }
#pragma unroll
  for (int s = 0; s < 2; s++)
#pragma unroll
    for (int t = 0; t < 2; t++) {
      int n = nb + t * 16 + l16;
      if (n < N) {
        float bv = bias[(long long)z * biasS + n];
#pragma unroll
        for (int r = 0; r < 8; r++) {
          int m = s * 16 + khalf * 8 + r;
          float v = acc[s][t][r] + bv;
          if constexpr (ACT == 1) v = fmaxf(v, 0.f);
          if constexpr (ACT == 2) v = (v > 0.f) ? v : 0.2f * v;
          Cm[(long long)z * Cs + (long long)m * N + n] = v;
        }
      }
    }
}

// ---------------------------------------------------------------------------
// Pooling / heads / GCN
// ---------------------------------------------------------------------------
__global__ void maxpool_kernel(const unsigned short* __restrict__ in,
                               unsigned short* __restrict__ out, int total) {
  int idx = blockIdx.x * 256 + threadIdx.x;  // [32][56][56][64]
  if (idx >= total) return;
  int c = idx & 63;
  int p = idx >> 6;
  int ow = p % 56; p /= 56;
  int oh = p % 56;
  int n = p / 56;
  float mx = -3.402823466e38f;
  for (int kh = 0; kh < 3; kh++)
    for (int kw = 0; kw < 3; kw++) {
      int ih = oh * 2 - 1 + kh, iw = ow * 2 - 1 + kw;
      if (ih >= 0 && ih < 112 && iw >= 0 && iw < 112)
        mx = fmaxf(mx, bf2f(in[(((long long)n * 112 + ih) * 112 + iw) * 64 + c]));
    }
  out[idx] = f2bf(mx);
}

__global__ void avgpool_kernel(const unsigned short* __restrict__ in,
                               unsigned short* __restrict__ g, int total) {
  int idx = blockIdx.x * 256 + threadIdx.x;  // [32][2048]
  if (idx >= total) return;
  int c = idx & 2047;
  int n = idx >> 11;
  float s = 0.f;
  for (int p = 0; p < 49; p++) s += bf2f(in[((long long)n * 49 + p) * 2048 + c]);
  g[idx] = f2bf(s * (1.f / 49.f));
}

__global__ void bn_feat_kernel(float* __restrict__ f, const float* g,
                               const float* b, const float* m, const float* v,
                               int total) {
  int i = blockIdx.x * 256 + threadIdx.x;  // feats [35][32][32]
  if (i >= total) return;
  int a = i >> 10;
  float s = g[a] * rsqrtf(v[a] + 1e-5f);
  f[i] = (f[i] - m[a]) * s + b[a];
}

__global__ void classifier_kernel(const float* __restrict__ feats,
                                  const float* __restrict__ W,
                                  const float* __restrict__ bb,
                                  float* __restrict__ out, int total) {
  int idx = blockIdx.x * 256 + threadIdx.x;
  if (idx >= total) return;
  int b = idx / 35, a = idx - b * 35;
  float s = bb[a];
  for (int d = 0; d < 32; d++) s += feats[(a * 32 + b) * 32 + d] * W[a * 32 + d];
  out[b * 35 + a] = s;
}

__global__ void build_adj_kernel(const int* __restrict__ adj,
                                 const float* __restrict__ ew, int E,
                                 float* __restrict__ Amat,
                                 float* __restrict__ deg) {
  int tid = threadIdx.x;  // single block
  for (int i = tid; i < 35 * 35; i += 256) Amat[i] = 0.f;
  for (int i = tid; i < 35; i += 256) deg[i] = 0.f;
  __syncthreads();
  for (int e = tid; e < E + 35; e += 256) {
    int d = (e < E) ? adj[E + e] : (e - E);
    float w = (e < E) ? ew[e] : 1.f;
    atomicAdd(&deg[d], w);
  }
  __threadfence();
  __syncthreads();
  for (int e = tid; e < E + 35; e += 256) {
    int s = (e < E) ? adj[e] : (e - E);
    int d = (e < E) ? adj[E + e] : (e - E);
    float w = (e < E) ? ew[e] : 1.f;
    float di = (deg[d] > 0.f) ? rsqrtf(deg[d]) : 0.f;
    float si = (deg[s] > 0.f) ? rsqrtf(deg[s]) : 0.f;
    atomicAdd(&Amat[d * 35 + s], si * w * di);
  }
}

__global__ void gcn_forward_kernel(const float* __restrict__ feats,
                                   const float* __restrict__ Amat,
                                   const float* __restrict__ W1,
                                   const float* __restrict__ b1,
                                   const float* __restrict__ W2,
                                   const float* __restrict__ b2,
                                   float* __restrict__ out) {
  __shared__ float T[35 * 32];
  __shared__ float Hh[35 * 32];
  __shared__ float t2[35];
  int b = blockIdx.x;
  int tid = threadIdx.x;
  for (int idx = tid; idx < 35 * 32; idx += 256) {
    int i = idx >> 5, f = idx & 31;
    float s = 0.f;
    for (int j = 0; j < 32; j++) s += feats[(i * 32 + b) * 32 + j] * W1[j * 32 + f];
    T[idx] = s;
  }
  __syncthreads();
  for (int idx = tid; idx < 35 * 32; idx += 256) {
    int i = idx >> 5, f = idx & 31;
    float s = b1[f];
    for (int j = 0; j < 35; j++) s += Amat[i * 35 + j] * T[j * 32 + f];
    Hh[idx] = (s > 0.f) ? s : 0.2f * s;
  }
  __syncthreads();
  for (int i = tid; i < 35; i += 256) {
    float s = 0.f;
    for (int f = 0; f < 32; f++) s += Hh[i * 32 + f] * W2[f];
    t2[i] = s;
  }
  __syncthreads();
  for (int i = tid; i < 35; i += 256) {
    float s = b2[0];
    for (int j = 0; j < 35; j++) s += Amat[i * 35 + j] * t2[j];
    out[b * 35 + i] = s;
  }
}

// ---------------------------------------------------------------------------
// Host orchestration
// ---------------------------------------------------------------------------
struct BnP { const float *g, *b, *m, *v; };

extern "C" void kernel_launch(void* const* d_in, const int* in_sizes, int n_in,
                              void* d_out, int out_size, void* d_ws,
                              size_t ws_size, hipStream_t stream) {
  (void)n_in; (void)out_size; (void)ws_size;
  const float* x    = (const float*)d_in[0];
  const int*   adjp = (const int*)d_in[1];
  const float* ewp  = (const float*)d_in[2];
  const int E = in_sizes[1] / 2;

  int cur = 3;
  auto nf  = [&]() { return (const float*)d_in[cur++]; };
  auto nbn = [&]() { BnP p; p.g = nf(); p.b = nf(); p.m = nf(); p.v = nf(); return p; };

  char* wsbase = (char*)d_ws;
  size_t wsoff = 0;
  auto walloc = [&](size_t bytes) -> void* {
    void* r = wsbase + wsoff;
    wsoff += (bytes + 255) & ~(size_t)255;
    return r;
  };
  auto lg2 = [](int v) { int s = 0; while ((1 << s) < v) s++; return s; };

  // Four ping-pong activation buffers (bf16 NHWC), largest = 32x112x112x64.
  const size_t ACT_ELEMS = (size_t)32 * 112 * 112 * 64;
  unsigned short* act[4];
  for (int i = 0; i < 4; i++) act[i] = (unsigned short*)walloc(ACT_ELEMS * 2);

  // Input pack fp32 NCHW -> bf16 NHWC (C 3->4).
  unsigned short* xbf = (unsigned short*)walloc((size_t)32 * 224 * 224 * 4 * 2);
  {
    int total = 32 * 224 * 224 * 4;
    pack_input_kernel<<<dim3((total + 255) / 256), dim3(256), 0, stream>>>(x, xbf, total);
  }

  auto run_conv = [&](const unsigned short* in, int IH, int IW, int ICp,
                      int srcIC, const float* w_oihw, int KH, int KW, int KWp,
                      int stride, int pad, BnP bn, const unsigned short* res,
                      int relu, unsigned short* out, int OC) {
    int OH = (IH + 2 * pad - KH) / stride + 1;
    int OW = (IW + 2 * pad - KW) / stride + 1;
    int Ktot = KH * KWp * ICp;
    unsigned short* wbf = (unsigned short*)walloc((size_t)OC * Ktot * 2);
    float* sc = (float*)walloc((size_t)OC * 4);
    float* bi = (float*)walloc((size_t)OC * 4);
    long long wtotal = (long long)OC * Ktot;
    conv_w_prep_kernel<<<dim3((unsigned)((wtotal + 255) / 256)), dim3(256), 0, stream>>>(
        w_oihw, wbf, srcIC, KH, KW, KWp, ICp, wtotal);
    bn_prep_kernel<<<dim3((OC + 255) / 256), dim3(256), 0, stream>>>(
        bn.g, bn.b, bn.m, bn.v, sc, bi, OC);
    ConvP cp;
    cp.inp = in; cp.wgt = wbf; cp.scale = sc; cp.bias = bi; cp.res = res;
    cp.outp = out;
    cp.IH = IH; cp.IW = IW; cp.icShift = lg2(ICp);
    cp.OH = OH; cp.OW = OW; cp.OC = OC; cp.KWp = KWp; cp.kwShift = lg2(KWp);
    cp.Ktot = Ktot;
    cp.stride = stride; cp.pad = pad;
    int M = 32 * OH * OW;
    dim3 grid(OC / 64, (M + 127) / 128);
    if (ICp < 32)
      conv_igemm_wmma<true, false, true><<<grid, dim3(256), 0, stream>>>(cp);
    else if (res)
      conv_igemm_wmma<false, true, true><<<grid, dim3(256), 0, stream>>>(cp);
    else if (relu)
      conv_igemm_wmma<false, false, true><<<grid, dim3(256), 0, stream>>>(cp);
    else
      conv_igemm_wmma<false, false, false><<<grid, dim3(256), 0, stream>>>(cp);
  };

  // ---- stem: 7x7/2 conv (K padded to 7x8x4 = 224) + BN/ReLU + maxpool -----
  const float* c1w = nf(); BnP c1bn = nbn();
  run_conv(xbf, 224, 224, 4, 3, c1w, 7, 7, 8, 2, 3, c1bn, nullptr, 1, act[0], 64);
  {
    int total = 32 * 56 * 56 * 64;
    maxpool_kernel<<<dim3((total + 255) / 256), dim3(256), 0, stream>>>(act[0], act[1], total);
  }

  // ---- bottleneck stages ---------------------------------------------------
  int ci = 1, H = 56, W = 56, in_ch = 64;
  const int PLANES[4] = {64, 128, 256, 512};
  const int NBLK[4]   = {3, 4, 6, 3};
  const int STR[4]    = {1, 2, 2, 2};
  for (int li = 0; li < 4; li++) {
    for (int j = 0; j < NBLK[li]; j++) {
      int s = (j == 0) ? STR[li] : 1;
      int planes = PLANES[li];
      const float* w1 = nf(); BnP bn1 = nbn();
      const float* w2 = nf(); BnP bn2 = nbn();
      const float* w3 = nf(); BnP bn3 = nbn();
      const float* dw = nullptr; BnP dbn = {};
      bool down = (j == 0);  // every stage's first block has a downsample path
      if (down) { dw = nf(); dbn = nbn(); }

      unsigned short* idt = act[ci];
      unsigned short* t1  = act[(ci + 1) & 3];
      unsigned short* t2  = act[(ci + 2) & 3];
      run_conv(idt, H, W, in_ch, in_ch, w1, 1, 1, 1, 1, 0, bn1, nullptr, 1, t1, planes);
      int H2 = (H + 2 - 3) / s + 1, W2 = (W + 2 - 3) / s + 1;
      run_conv(t1, H, W, planes, planes, w2, 3, 3, 3, s, 1, bn2, nullptr, 1, t2, planes);
      const unsigned short* resid = idt;
      if (down) {
        unsigned short* db = act[(ci + 3) & 3];
        run_conv(idt, H, W, in_ch, in_ch, dw, 1, 1, 1, s, 0, dbn, nullptr, 0, db, planes * 4);
        resid = db;
      }
      unsigned short* ob = act[(ci + 1) & 3];  // t1 is dead now
      run_conv(t2, H2, W2, planes, planes, w3, 1, 1, 1, 1, 0, bn3, resid, 1, ob, planes * 4);
      ci = (ci + 1) & 3; H = H2; W = W2; in_ch = planes * 4;
    }
  }

  // ---- heads ---------------------------------------------------------------
  unsigned short* featT = act[ci];  // [32][7][7][2048] bf16
  unsigned short* gbf = (unsigned short*)walloc((size_t)32 * 2048 * 2);
  {
    int total = 32 * 2048;
    avgpool_kernel<<<dim3((total + 255) / 256), dim3(256), 0, stream>>>(featT, gbf, total);
  }

  const float* W1 = nf(); const float* b1 = nf();
  const float* W2 = nf(); const float* b2 = nf();
  BnP bnf = nbn();
  const float* clsW = nf(); const float* clsb = nf();
  const float* g1W = nf(); const float* g1b = nf();
  const float* g2W = nf(); const float* g2b = nf();

  unsigned short* W1t = (unsigned short*)walloc((size_t)35 * 1024 * 2048 * 2);
  unsigned short* W2t = (unsigned short*)walloc((size_t)35 * 32 * 1024 * 2);
  {
    long long t1 = 35LL * 1024 * 2048;
    gemm_w_prep_kernel<<<dim3((unsigned)((t1 + 255) / 256)), dim3(256), 0, stream>>>(W1, W1t, 2048, 1024, t1);
    long long t2 = 35LL * 32 * 1024;
    gemm_w_prep_kernel<<<dim3((unsigned)((t2 + 255) / 256)), dim3(256), 0, stream>>>(W2, W2t, 1024, 32, t2);
  }

  float* H1 = (float*)walloc((size_t)35 * 32 * 1024 * 4);
  unsigned short* H1b = (unsigned short*)walloc((size_t)35 * 32 * 1024 * 2);
  float* feats = (float*)walloc((size_t)35 * 32 * 32 * 4);

  // MLP layer 1: 35x (32x2048 @ 2048x1024), leaky-ReLU
  gemm_bf16_m32<2><<<dim3(4, 35), dim3(256), 0, stream>>>(
      gbf, 0LL, W1t, 2048LL * 1024, b1, 1024LL, H1, 32LL * 1024, 2048, 1024, 0);
  {
    long long n = 35LL * 32 * 1024;
    cvt_bf16_kernel<<<dim3((unsigned)((n + 255) / 256)), dim3(256), 0, stream>>>(H1, H1b, n);
  }
  // MLP layer 2: 35x (32x1024 @ 1024x32), leaky-ReLU
  gemm_bf16_m32<2><<<dim3(1, 35), dim3(256), 0, stream>>>(
      H1b, 32LL * 1024, W2t, 32LL * 1024, b2, 32LL, feats, 32LL * 32, 1024, 32, 0);
  {
    int n = 35 * 32 * 32;
    bn_feat_kernel<<<dim3((n + 255) / 256), dim3(256), 0, stream>>>(
        feats, bnf.g, bnf.b, bnf.m, bnf.v, n);
  }

  float* outF = (float*)d_out;  // [0:1120)=gcn out, [1120:2240)=feature_outputs
  classifier_kernel<<<dim3(5), dim3(256), 0, stream>>>(feats, clsW, clsb, outF + 1120, 1120);

  float* Amat = (float*)walloc(35 * 35 * 4);
  float* deg  = (float*)walloc(35 * 4);
  build_adj_kernel<<<dim3(1), dim3(256), 0, stream>>>(adjp, ewp, E, Amat, deg);
  gcn_forward_kernel<<<dim3(32), dim3(256), 0, stream>>>(feats, Amat, g1W, g1b, g2W, g2b, outF);
}
